// GNTransformer_joint2_12111807775413
// MI455X (gfx1250) — matrix operand order repaired and checked by reading, not compiled, not run on hardware
//
#include <hip/hip_runtime.h>
#include <hip/hip_bf16.h>
#include <math.h>

#define NND 20000
#define NED 320000
#define BB  32
#define HH  128
#define NEQ 201
#define NPGD 625
#define FFND 512

typedef __attribute__((ext_vector_type(16))) __bf16 bf16x16;
typedef __attribute__((ext_vector_type(8)))  float  f32x8;

#if defined(__AMDGCN__) &&                                             \
    __has_builtin(__builtin_amdgcn_global_load_async_to_lds_b32) &&    \
    __has_builtin(__builtin_amdgcn_global_load_async_to_lds_b128) &&   \
    __has_builtin(__builtin_amdgcn_s_wait_asynccnt)
#define USE_ASYNC_LDS 1
#else
#define USE_ASYNC_LDS 0
#endif

#if USE_ASYNC_LDS
typedef int v4i_pack __attribute__((vector_size(4 * sizeof(int))));
typedef __attribute__((address_space(1))) v4i_pack gas_v4i;
typedef __attribute__((address_space(3))) v4i_pack las_v4i;
#endif

// async (or fallback sync) copy of 16 bytes global -> LDS (16B aligned)
__device__ inline void stage4_f32(const float* g, float* l) {
#if USE_ASYNC_LDS
  __builtin_amdgcn_global_load_async_to_lds_b128((gas_v4i*)g, (las_v4i*)l, 0, 0);
#else
  *(float4*)l = *(const float4*)g;
#endif
}
__device__ inline void stage_wait1() {
#if USE_ASYNC_LDS
  __builtin_amdgcn_s_wait_asynccnt(1);
#endif
}
__device__ inline void stage_wait0() {
#if USE_ASYNC_LDS
  __builtin_amdgcn_s_wait_asynccnt(0);
#endif
}

__device__ inline f32x8 wmma_bf(bf16x16 a, bf16x16 b, f32x8 c) {
  return __builtin_amdgcn_wmma_f32_16x16x32_bf16(false, a, false, b, (short)0, c,
                                                 false, false);
}

// A fragment: 16x32 bf16 (contiguous k = 16*(lane/16) + e). f(m,k) -> float
template <typename F>
__device__ inline bf16x16 load_afrag(F f) {
  const int lane = threadIdx.x & 31;
  const int m = lane & 15;
  const int h8 = (lane >> 4) << 4;
  bf16x16 a;
#pragma unroll
  for (int r = 0; r < 8; ++r) {
    const int kb = h8 + 2 * r;
    a[2 * r]     = (__bf16)f(m, kb);
    a[2 * r + 1] = (__bf16)f(m, kb + 1);
  }
  return a;
}

// B fragment from float source (activations). f(k,n)
template <typename F>
__device__ inline bf16x16 load_bfrag(F f) {
  const int lane = threadIdx.x & 31;
  const int n = lane & 15;
  const int kg = (lane >> 4) << 4;
  bf16x16 b;
#pragma unroll
  for (int r = 0; r < 8; ++r) {
    b[2 * r]     = (__bf16)f(kg + 2 * r, n);
    b[2 * r + 1] = (__bf16)f(kg + 2 * r + 1, n);
  }
  return b;
}

// B fragment from pre-swizzled packed weights: one 32B load per lane.
__device__ inline bf16x16 load_pw(const __bf16* __restrict__ P, int kt, int nt,
                                  int ntN) {
  const int lane = threadIdx.x & 31;
  const bf16x16* p = (const bf16x16*)P;
  return p[((size_t)(kt * ntN + nt) << 5) + lane];
}

// pack W[K,N] fp32 -> bf16 fragment-order: [(kt*ntN+nt)][lane][16 elems]
// element e of lane corresponds to k = kt*32 + (lane>=16)*16 + e, n = nt*16+lane%16
__global__ void pack_w_kernel(const float* __restrict__ W,
                              __bf16* __restrict__ P, int K, int N,
                              size_t total) {
  const size_t i = (size_t)blockIdx.x * 256 + threadIdx.x;
  if (i >= total) return;
  const int ntN = N >> 4;
  const int e = (int)(i & 15);
  const int lane = (int)((i >> 4) & 31);
  const size_t tile = i >> 9;
  const int nt = (int)(tile % ntN);
  const int kt = (int)(tile / ntN);
  const int k = kt * 32 + ((lane >> 4) << 4) + e;
  const int n = nt * 16 + (lane & 15);
  P[i] = (k < K) ? (__bf16)W[(size_t)k * N + n] : (__bf16)0.f;
}

// ---------------- generic WMMA GEMM: out = [res +] act((A@W + bias) * scale)
// Requirements (all call sites satisfy): M%16==0, K%32==0, N%128==0, Ka%32==0.
// Block = 128 threads; each wave computes two 16x16 C tiles (block: 16x128).
template <bool CAT2>
__global__ void wmma_gemm_kernel(const float* __restrict__ A0,
                                 const float* __restrict__ A1, int Ka,
                                 const __bf16* __restrict__ PW,
                                 const float* __restrict__ bias,
                                 float* __restrict__ out, int M, int K, int N,
                                 float scale, int relu, int residual) {
  __shared__ float ldsA[2][16][36];
  const int tid = threadIdx.x;
  const int wv = tid >> 5;
  const int lane = tid & 31;
  const int m0 = blockIdx.x * 16;
  const int n0 = blockIdx.y * 128 + wv * 32;
  const int nt0 = n0 >> 4;
  const int ntN = N >> 4;
  const int rT = tid >> 3;         // staged row
  const int cT = (tid & 7) * 4;    // staged col group (4 floats)

  auto gaddr = [&](int k0) -> const float* {
    const int m = m0 + rT;
    const int k = k0 + cT;
    if (CAT2)
      return (k < Ka) ? &A0[(size_t)m * Ka + k]
                      : &A1[(size_t)m * (K - Ka) + (k - Ka)];
    return &A0[(size_t)m * K + k];
  };
  auto issue = [&](int t) { stage4_f32(gaddr(t * 32), &ldsA[t & 1][rT][cT]); };

  const int nt = K >> 5;
  const f32x8 zero = {0.f, 0.f, 0.f, 0.f, 0.f, 0.f, 0.f, 0.f};
  f32x8 acc0 = zero, acc1 = zero;
  issue(0);
  issue(1);
  for (int t = 0; t < nt; ++t) {
    if (t + 1 < nt) stage_wait1(); else stage_wait0();
    __syncthreads();
    if (t + 2 < nt)
      __builtin_prefetch(
          (const void*)((const bf16x16*)PW + ((size_t)((t + 2) * ntN + nt0) << 5)),
          0, 3);
    bf16x16 a = load_afrag([&](int mm, int kk) { return ldsA[t & 1][mm][kk]; });
    acc0 = wmma_bf(a, load_pw(PW, t, nt0, ntN), acc0);
    acc1 = wmma_bf(a, load_pw(PW, t, nt0 + 1, ntN), acc1);
    __syncthreads();
    if (t + 2 < nt) issue(t + 2);
  }
  const int n = lane & 15;
  const int mb = (lane >> 4) << 3;
  const float bs0 = bias ? bias[n0 + n] : 0.f;
  const float bs1 = bias ? bias[n0 + 16 + n] : 0.f;
#pragma unroll
  for (int r = 0; r < 8; ++r) {
    const int m = m0 + mb + r;
    float v0 = (acc0[r] + bs0) * scale;
    float v1 = (acc1[r] + bs1) * scale;
    if (relu) { v0 = fmaxf(v0, 0.f); v1 = fmaxf(v1, 0.f); }
    const size_t o = (size_t)m * N + n0 + n;
    if (residual) { out[o] += v0; out[o + 16] += v1; }
    else          { out[o] = v0;  out[o + 16] = v1; }
  }
}

// ---------------- encoder MLP2: out = prelu(A@W1+b1)@W2+b2, hidden=128
__global__ void mlp2_plain_kernel(const float* __restrict__ A,
                                  const __bf16* __restrict__ PW1,
                                  const float* __restrict__ b1,
                                  const float* __restrict__ palpha,
                                  const __bf16* __restrict__ PW2,
                                  const float* __restrict__ b2,
                                  float* __restrict__ out, int K1) {
  __shared__ float ldsA[16][33];
  __shared__ float hid[16][129];
  const int tid = threadIdx.x;
  const int wv = tid >> 5;
  const int lane = tid & 31;
  const int m0 = blockIdx.x * 16;
  const f32x8 zero = {0.f, 0.f, 0.f, 0.f, 0.f, 0.f, 0.f, 0.f};
  f32x8 acc[2] = {zero, zero};
  for (int k0 = 0; k0 < K1; k0 += 32) {
#pragma unroll
    for (int i = 0; i < 4; ++i) {
      const int el = tid + i * 128;
      const int mr = el >> 5, kc = el & 31;
      const int k = k0 + kc;
      ldsA[mr][kc] = (k < K1) ? A[(size_t)(m0 + mr) * K1 + k] : 0.f;
    }
    __syncthreads();
    bf16x16 a = load_afrag([&](int mm, int kk) { return ldsA[mm][kk]; });
#pragma unroll
    for (int j = 0; j < 2; ++j)
      acc[j] = wmma_bf(a, load_pw(PW1, k0 >> 5, wv * 2 + j, 8), acc[j]);
    __syncthreads();
  }
  {
    const float alpha = palpha[0];
    const int n = lane & 15;
    const int mb = (lane >> 4) << 3;
#pragma unroll
    for (int j = 0; j < 2; ++j) {
      const int n0 = (wv * 2 + j) * 16;
#pragma unroll
      for (int r = 0; r < 8; ++r) {
        float v = acc[j][r] + b1[n0 + n];
        hid[mb + r][n0 + n] = (v >= 0.f) ? v : alpha * v;
      }
    }
  }
  __syncthreads();
  f32x8 acc2[2] = {zero, zero};
  for (int k0 = 0; k0 < HH; k0 += 32) {
    bf16x16 a = load_afrag([&](int mm, int kk) { return hid[mm][k0 + kk]; });
#pragma unroll
    for (int j = 0; j < 2; ++j)
      acc2[j] = wmma_bf(a, load_pw(PW2, k0 >> 5, wv * 2 + j, 8), acc2[j]);
  }
  {
    const int n = lane & 15;
    const int mb = (lane >> 4) << 3;
#pragma unroll
    for (int j = 0; j < 2; ++j) {
      const int n0 = (wv * 2 + j) * 16;
#pragma unroll
      for (int r = 0; r < 8; ++r)
        out[(size_t)(m0 + mb + r) * HH + n0 + n] = acc2[j][r] + b2[n0 + n];
    }
  }
}

// ---------------- fused GNN layer (edge or node):
// h = prelu(LN(in@W1+b1)); d = h@W2+b2; io += d; (edge) agg[col] += d atomically
// K1 = 384 (edge: [x[row]|x[col]|e]) or 256 (node: [x|agg]); hidden = 256.
__global__ void fused_gnn_layer_kernel(
    const float* __restrict__ xsrc, const float* __restrict__ aggsrc,
    const int* __restrict__ erow, const int* __restrict__ ecol,
    const __bf16* __restrict__ PW1, const float* __restrict__ b1,
    const float* __restrict__ lng, const float* __restrict__ lnb,
    const float* __restrict__ palpha, const __bf16* __restrict__ PW2,
    const float* __restrict__ b2, float* __restrict__ io,
    float* __restrict__ agg, int K1) {
  __shared__ float ldsA[2][16][36];
  __shared__ float hid[16][257];
  __shared__ float redS[16][8];
  __shared__ float redQ[16][8];
  __shared__ float mu[16], rs[16];
  const int tid = threadIdx.x;
  const int wv = tid >> 5;
  const int lane = tid & 31;
  const int m0 = blockIdx.x * 16;
  const int rT = tid >> 3;
  const int cT = (tid & 7) * 4;
  const int edgeMode = (erow != nullptr);

  // hoisted row base pointers for the staged row (one per 128-wide K segment)
  const float *s0, *s1, *s2;
  {
    const int m = m0 + rT;
    if (edgeMode) {
      s0 = xsrc + (size_t)erow[m] * HH;
      s1 = xsrc + (size_t)ecol[m] * HH;
      s2 = io + (size_t)m * HH;
    } else {
      s0 = io + (size_t)m * HH;
      s1 = aggsrc + (size_t)m * HH;
      s2 = nullptr;
    }
  }
  auto issue = [&](int t) {
    const int k0 = t * 32;
    const int sg = k0 >> 7;
    const float* s = (sg == 0) ? s0 : (sg == 1) ? s1 : s2;
    stage4_f32(s + (k0 & 127) + cT, &ldsA[t & 1][rT][cT]);
  };

  const int nt = K1 >> 5;  // 12 or 8
  const f32x8 zero = {0.f, 0.f, 0.f, 0.f, 0.f, 0.f, 0.f, 0.f};
  f32x8 acc[4] = {zero, zero, zero, zero};
  issue(0);
  issue(1);
  for (int t = 0; t < nt; ++t) {
    if (t + 1 < nt) stage_wait1(); else stage_wait0();
    __syncthreads();
    bf16x16 a = load_afrag([&](int mm, int kk) { return ldsA[t & 1][mm][kk]; });
#pragma unroll
    for (int j = 0; j < 4; ++j)
      acc[j] = wmma_bf(a, load_pw(PW1, t, wv * 4 + j, 16), acc[j]);
    __syncthreads();
    if (t + 2 < nt) issue(t + 2);
  }
  {
    const int n = lane & 15;
    const int mb = (lane >> 4) << 3;
#pragma unroll
    for (int j = 0; j < 4; ++j) {
      const int n0 = (wv * 4 + j) * 16;
#pragma unroll
      for (int r = 0; r < 8; ++r)
        hid[mb + r][n0 + n] = acc[j][r] + b1[n0 + n];
    }
  }
  __syncthreads();
  {  // LayerNorm(256) partial sums: 8 threads per row
    const int r = tid >> 3, t = tid & 7;
    float s = 0.f, q = 0.f;
    for (int c = t * 32; c < t * 32 + 32; ++c) {
      const float v = hid[r][c];
      s += v; q += v * v;
    }
    redS[r][t] = s; redQ[r][t] = q;
  }
  __syncthreads();
  if ((tid & 7) == 0) {
    const int r = tid >> 3;
    float s = 0.f, q = 0.f;
    for (int t = 0; t < 8; ++t) { s += redS[r][t]; q += redQ[r][t]; }
    const float m = s * (1.f / 256.f);
    const float var = q * (1.f / 256.f) - m * m;
    mu[r] = m; rs[r] = rsqrtf(var + 1e-5f);
  }
  __syncthreads();
  {
    const float alpha = palpha[0];
    const int r = tid >> 3, t = tid & 7;
    const float m = mu[r], rv = rs[r];
    for (int c = t * 32; c < t * 32 + 32; ++c) {
      float v = (hid[r][c] - m) * rv * lng[c] + lnb[c];
      hid[r][c] = (v >= 0.f) ? v : alpha * v;
    }
  }
  __syncthreads();
  f32x8 acc2[2] = {zero, zero};
  for (int k0 = 0; k0 < 256; k0 += 32) {
    bf16x16 a = load_afrag([&](int mm, int kk) { return hid[mm][k0 + kk]; });
#pragma unroll
    for (int j = 0; j < 2; ++j)
      acc2[j] = wmma_bf(a, load_pw(PW2, k0 >> 5, wv * 2 + j, 8), acc2[j]);
  }
  {
    const int n = lane & 15;
    const int mb = (lane >> 4) << 3;
#pragma unroll
    for (int j = 0; j < 2; ++j) {
      const int n0 = (wv * 2 + j) * 16;
#pragma unroll
      for (int r = 0; r < 8; ++r) {
        const int m = m0 + mb + r;
        const float v = acc2[j][r] + b2[n0 + n];
        io[(size_t)m * HH + n0 + n] += v;
        if (agg) atomicAdd(&agg[(size_t)ecol[m] * HH + n0 + n], v);
      }
    }
  }
}

// ---------------- attention: logits[b,q,p] = sum_k qh[q,b,k]*kh[p,b,k]
__global__ void attn_score_kernel(const float* __restrict__ qh,
                                  const float* __restrict__ kh,
                                  float* __restrict__ logits) {
  __shared__ float ldsQ[16][33];
  const int tid = threadIdx.x;
  const int wv = tid >> 5;
  const int lane = tid & 31;
  const int q0 = blockIdx.x * 16;
  const int p0 = (blockIdx.y * 4 + wv) * 16;
  const int b = blockIdx.z;
  f32x8 acc = {0.f, 0.f, 0.f, 0.f, 0.f, 0.f, 0.f, 0.f};
  for (int k0 = 0; k0 < HH; k0 += 32) {
#pragma unroll
    for (int i = 0; i < 4; ++i) {
      const int el = tid + i * 128;
      const int mr = el >> 5, kc = el & 31;
      const int q = q0 + mr;
      ldsQ[mr][kc] = (q < NEQ) ? qh[((size_t)q * BB + b) * HH + k0 + kc] : 0.f;
    }
    __syncthreads();
    bf16x16 a = load_afrag([&](int mm, int kk) { return ldsQ[mm][kk]; });
    bf16x16 bfr = load_bfrag([&](int kk, int nn) {
      const int p = p0 + nn;
      return (p < NPGD) ? kh[((size_t)p * BB + b) * HH + k0 + kk] : 0.f;
    });
    acc = wmma_bf(a, bfr, acc);
    __syncthreads();
  }
  const int n = lane & 15;
  const int mb = (lane >> 4) << 3;
#pragma unroll
  for (int r = 0; r < 8; ++r) {
    const int q = q0 + mb + r, p = p0 + n;
    if (q < NEQ && p < NPGD)
      logits[((size_t)b * NEQ + q) * NPGD + p] = acc[r];
  }
}

__global__ void softmax_kernel(float* __restrict__ logits) {
  __shared__ float red[128];
  const int tid = threadIdx.x;
  float* rowp = logits + (size_t)blockIdx.x * NPGD;
  float mx = -1e30f;
  for (int i = tid; i < NPGD; i += 128) mx = fmaxf(mx, rowp[i]);
  red[tid] = mx; __syncthreads();
  for (int s = 64; s > 0; s >>= 1) {
    if (tid < s) red[tid] = fmaxf(red[tid], red[tid + s]);
    __syncthreads();
  }
  mx = red[0]; __syncthreads();
  float sum = 0.f;
  for (int i = tid; i < NPGD; i += 128) {
    const float e = expf(rowp[i] - mx);
    rowp[i] = e; sum += e;
  }
  red[tid] = sum; __syncthreads();
  for (int s = 64; s > 0; s >>= 1) {
    if (tid < s) red[tid] += red[tid + s];
    __syncthreads();
  }
  const float inv = 1.f / red[0];
  for (int i = tid; i < NPGD; i += 128) rowp[i] *= inv;
}

// out[q,b,:] = sum_p attn[b,q,p] * vh[p,b,:]
__global__ void attn_apply_kernel(const float* __restrict__ attn,
                                  const float* __restrict__ vh,
                                  float* __restrict__ outp) {
  __shared__ float ldsA[16][33];
  const int tid = threadIdx.x;
  const int wv = tid >> 5;
  const int lane = tid & 31;
  const int q0 = blockIdx.x * 16;
  const int n0 = (blockIdx.y * 4 + wv) * 16;
  const int b = blockIdx.z;
  f32x8 acc = {0.f, 0.f, 0.f, 0.f, 0.f, 0.f, 0.f, 0.f};
  for (int k0 = 0; k0 < NPGD; k0 += 32) {
#pragma unroll
    for (int i = 0; i < 4; ++i) {
      const int el = tid + i * 128;
      const int mr = el >> 5, kc = el & 31;
      const int q = q0 + mr, k = k0 + kc;
      ldsA[mr][kc] = (q < NEQ && k < NPGD)
                         ? attn[((size_t)b * NEQ + q) * NPGD + k] : 0.f;
    }
    __syncthreads();
    bf16x16 a = load_afrag([&](int mm, int kk) { return ldsA[mm][kk]; });
    bf16x16 bfr = load_bfrag([&](int kk, int nn) {
      const int k = k0 + kk;
      return (k < NPGD) ? vh[((size_t)k * BB + b) * HH + n0 + nn] : 0.f;
    });
    acc = wmma_bf(a, bfr, acc);
    __syncthreads();
  }
  const int n = lane & 15;
  const int mb = (lane >> 4) << 3;
#pragma unroll
  for (int r = 0; r < 8; ++r) {
    const int q = q0 + mb + r;
    if (q < NEQ) outp[((size_t)q * BB + b) * HH + n0 + n] = acc[r];
  }
}

// ---------------- small kernels ----------------
__global__ void ln_kernel(const float* __restrict__ in,
                          const float* __restrict__ g,
                          const float* __restrict__ bt,
                          float* __restrict__ outp) {
  __shared__ float s1[128], s2[128];
  const int tid = threadIdx.x;
  const size_t row = blockIdx.x;
  const float v = in[row * HH + tid];
  s1[tid] = v; s2[tid] = v * v;
  __syncthreads();
  for (int s = 64; s > 0; s >>= 1) {
    if (tid < s) { s1[tid] += s1[tid + s]; s2[tid] += s2[tid + s]; }
    __syncthreads();
  }
  const float m = s1[0] * (1.f / HH);
  const float var = s2[0] * (1.f / HH) - m * m;
  outp[row * HH + tid] = (v - m) * rsqrtf(var + 1e-5f) * g[tid] + bt[tid];
}

__device__ inline float sinusoid_val(int pos, int d) {
  const float c = -logf(10000.f) / 63.f;
  const int i = (d < 64) ? d : d - 64;
  const float ang = (float)pos * expf((float)i * c);
  return (d < 64) ? sinf(ang) : cosf(ang);
}

__global__ void seed_q_kernel(const float* __restrict__ embed,
                              float* __restrict__ q) {
  const size_t i = (size_t)blockIdx.x * 256 + threadIdx.x;
  if (i >= (size_t)NEQ * BB * HH) return;
  const int h = (int)(i % HH);
  const int s = (int)(i / HH) / BB;
  q[i] = embed[(size_t)s * HH + h] * 11.313708499f + sinusoid_val(s, h);
}

__global__ void seed_kv_kernel(const float* __restrict__ x,
                               float* __restrict__ kv) {
  const size_t i = (size_t)blockIdx.x * 256 + threadIdx.x;
  if (i >= (size_t)NPGD * BB * HH) return;
  const int h = (int)(i % HH);
  const int pb = (int)(i / HH);
  const int b = pb % BB, p = pb / BB;
  kv[i] = x[((size_t)b * NPGD + p) * HH + h] * 11.313708499f +
          sinusoid_val(p, h);
}

__global__ void gsum_kernel(const float* __restrict__ x, float* __restrict__ gs) {
  const int b = blockIdx.x, t = threadIdx.x;
  float s = 0.f;
  for (int p = 0; p < NPGD; ++p) s += x[((size_t)b * NPGD + p) * HH + t];
  gs[b * HH + t] = s;
}

__global__ void di_kernel(const float* __restrict__ en,
                          const float* __restrict__ graph,
                          const float* __restrict__ palpha,
                          float* __restrict__ di) {
  const size_t i = (size_t)blockIdx.x * 256 + threadIdx.x;
  if (i >= (size_t)NEQ * BB * HH) return;
  const int h = (int)(i % HH);
  const int b = ((int)(i / HH)) % BB;
  di[i] = en[i] + palpha[0] * graph[b * HH + h];
}

__global__ void prelu_kernel(float* __restrict__ y,
                             const float* __restrict__ pa, size_t n) {
  const size_t i = (size_t)blockIdx.x * 256 + threadIdx.x;
  if (i < n) { const float v = y[i]; y[i] = (v >= 0.f) ? v : pa[0] * v; }
}

__global__ void kzero(float* __restrict__ p, size_t n) {
  const size_t i = (size_t)blockIdx.x * 256 + threadIdx.x;
  if (i < n) p[i] = 0.f;
}

__global__ void dos_kernel(const float* __restrict__ hb,
                           const float* __restrict__ w,
                           const float* __restrict__ bs,
                           float* __restrict__ dos) {
  __shared__ float red[128];
  const int row = blockIdx.x, tid = threadIdx.x;
  red[tid] = hb[(size_t)row * HH + tid] * w[tid];
  __syncthreads();
  for (int s = 64; s > 0; s >>= 1) {
    if (tid < s) red[tid] += red[tid + s];
    __syncthreads();
  }
  if (tid == 0) {
    const int b = row % BB, s = row / BB;
    dos[b * NEQ + s] = red[0] + bs[0];
  }
}

__global__ void spark_kernel(const float* __restrict__ en,
                             const float* __restrict__ w,
                             const float* __restrict__ bs,
                             float* __restrict__ sp) {
  __shared__ float red[128];
  const int row = blockIdx.x, tid = threadIdx.x;  // row = s*32+b, s<200
  red[tid] = en[(size_t)row * HH + tid] * w[tid];
  __syncthreads();
  for (int s = 64; s > 0; s >>= 1) {
    if (tid < s) red[tid] += red[tid + s];
    __syncthreads();
  }
  if (tid == 0) sp[row] = 1.f / (1.f + expf(-(red[0] + bs[0])));
}

// ---------------- host side ----------------
// params flattened after [x, edge_attr, glob, edge_index, batch], sorted-key order
enum {
  I_alpha = 0, I_dec_b = 1, I_dec_w = 2, I_embed = 3,
  I_ee_w1 = 4, I_ee_b1 = 5, I_ee_a = 6, I_ee_w2 = 7, I_ee_b2 = 8,
  I_eg_w1 = 9, I_eg_b1 = 10, I_eg_a = 11, I_eg_w2 = 12, I_eg_b2 = 13,
  I_en_w1 = 14, I_en_b1 = 15, I_en_a = 16, I_en_w2 = 17, I_en_b2 = 18,
  I_oa = 19, I_ob1 = 20, I_ob2 = 21, I_olb = 22, I_olg = 23, I_ow1 = 24,
  I_ow2 = 25,
  I_proc0 = 26,  // +14/layer: ea,eb1,eb2,elb,elg,ew1,ew2,na,nb1,nb2,nlb,nlg,nw1,nw2
  I_sb = 68, I_sw = 69,
  I_tr0 = 70,    // +16/layer: bk,bo,bq,bv,f1,f2,fb1,fb2,ln0b,ln0g,ln1b,ln1g,wk,wo,wq,wv
  I_tr_lnb = 102, I_tr_lng = 103
};

extern "C" void kernel_launch(void* const* d_in, const int* in_sizes, int n_in,
                              void* d_out, int out_size, void* d_ws,
                              size_t ws_size, hipStream_t stream) {
  (void)in_sizes; (void)n_in; (void)out_size; (void)ws_size;
  const float* in_x    = (const float*)d_in[0];
  const float* in_ea   = (const float*)d_in[1];
  const float* in_glob = (const float*)d_in[2];
  const int*   ei      = (const int*)d_in[3];
  auto Pf = [&](int i) { return (const float*)d_in[5 + i]; };
  float* out = (float*)d_out;

  float* wp = (float*)d_ws;
  size_t off = 0;
  auto alloc = [&](size_t n) { float* p = wp + off; off += n; return p; };
  float* xbuf   = alloc((size_t)NND * HH);
  float* ebuf   = alloc((size_t)NED * HH);
  float* agg    = alloc((size_t)NND * HH);
  float* ubuf   = alloc((size_t)BB * HH);
  float* qbuf   = alloc((size_t)NEQ * BB * HH);
  float* kv     = alloc((size_t)NPGD * BB * HH);
  float* qn     = alloc((size_t)NEQ * BB * HH);
  float* kn     = alloc((size_t)NPGD * BB * HH);
  float* qh     = alloc((size_t)NEQ * BB * HH);
  float* kh     = alloc((size_t)NPGD * BB * HH);
  float* vh     = alloc((size_t)NPGD * BB * HH);
  float* logits = alloc((size_t)BB * NEQ * NPGD);
  float* aout   = alloc((size_t)NEQ * BB * HH);
  float* ffh    = alloc((size_t)NEQ * BB * FFND);
  float* tmp    = alloc((size_t)NEQ * BB * HH);
  float* enf    = alloc((size_t)NEQ * BB * HH);
  float* gsb    = alloc((size_t)BB * HH);
  float* grb    = alloc((size_t)BB * HH);
  float* dib    = alloc((size_t)NEQ * BB * HH);
  float* hb     = alloc((size_t)NEQ * BB * HH);

  // packed-weight allocations (bf16, fragment order, 64B aligned)
  auto allocPW = [&](int K, int N) -> __bf16* {
    const int KT = (K + 31) / 32, NT = N / 16;
    size_t fl = ((size_t)KT * NT * 512 + 1) / 2;
    fl = (fl + 15) & ~(size_t)15;
    return (__bf16*)alloc(fl);
  };
  auto packW = [&](const float* W, __bf16* P, int K, int N) {
    const int KT = (K + 31) / 32, NT = N / 16;
    const size_t total = (size_t)KT * NT * 512;
    pack_w_kernel<<<(int)((total + 255) / 256), 256, 0, stream>>>(W, P, K, N,
                                                                  total);
  };

  __bf16* p_en1 = allocPW(92, HH);
  __bf16* p_en2 = allocPW(HH, HH);
  __bf16* p_ee1 = allocPW(41, HH);
  __bf16* p_ee2 = allocPW(HH, HH);
  __bf16* p_eg1 = allocPW(2, HH);
  __bf16* p_eg2 = allocPW(HH, HH);
  __bf16 *p_ew1[3], *p_ew2[3], *p_nw1[3], *p_nw2[3];
  for (int l = 0; l < 3; ++l) {
    p_ew1[l] = allocPW(3 * HH, 2 * HH);
    p_ew2[l] = allocPW(2 * HH, HH);
    p_nw1[l] = allocPW(2 * HH, 2 * HH);
    p_nw2[l] = allocPW(2 * HH, HH);
  }
  __bf16 *p_wq[2], *p_wk[2], *p_wv[2], *p_wo[2], *p_f1[2], *p_f2[2];
  for (int t = 0; t < 2; ++t) {
    p_wq[t] = allocPW(HH, HH);
    p_wk[t] = allocPW(HH, HH);
    p_wv[t] = allocPW(HH, HH);
    p_wo[t] = allocPW(HH, HH);
    p_f1[t] = allocPW(HH, FFND);
    p_f2[t] = allocPW(FFND, HH);
  }
  __bf16* p_dec = allocPW(2 * HH, HH);
  __bf16* p_ow1 = allocPW(HH, HH);

  // pack all weights (cheap, amortized over thousands of M-tiles)
  packW(Pf(I_en_w1), p_en1, 92, HH);
  packW(Pf(I_en_w2), p_en2, HH, HH);
  packW(Pf(I_ee_w1), p_ee1, 41, HH);
  packW(Pf(I_ee_w2), p_ee2, HH, HH);
  packW(Pf(I_eg_w1), p_eg1, 2, HH);
  packW(Pf(I_eg_w2), p_eg2, HH, HH);
  for (int l = 0; l < 3; ++l) {
    const int PB = I_proc0 + 14 * l;
    packW(Pf(PB + 5), p_ew1[l], 3 * HH, 2 * HH);
    packW(Pf(PB + 6), p_ew2[l], 2 * HH, HH);
    packW(Pf(PB + 12), p_nw1[l], 2 * HH, 2 * HH);
    packW(Pf(PB + 13), p_nw2[l], 2 * HH, HH);
  }
  for (int t = 0; t < 2; ++t) {
    const int TB = I_tr0 + 16 * t;
    packW(Pf(TB + 14), p_wq[t], HH, HH);
    packW(Pf(TB + 12), p_wk[t], HH, HH);
    packW(Pf(TB + 15), p_wv[t], HH, HH);
    packW(Pf(TB + 13), p_wo[t], HH, HH);
    packW(Pf(TB + 4), p_f1[t], HH, FFND);
    packW(Pf(TB + 5), p_f2[t], FFND, HH);
  }
  packW(Pf(I_dec_w), p_dec, 2 * HH, HH);
  packW(Pf(I_ow1), p_ow1, HH, HH);

  const int MQ = NEQ * BB;    // 6432
  const int MK = NPGD * BB;   // 20000
  const float scaling = 0.0883883476483184f;  // H^-0.5

  // encoders
  mlp2_plain_kernel<<<NND / 16, 128, 0, stream>>>(
      in_x, p_en1, Pf(I_en_b1), Pf(I_en_a), p_en2, Pf(I_en_b2), xbuf, 92);
  mlp2_plain_kernel<<<NED / 16, 128, 0, stream>>>(
      in_ea, p_ee1, Pf(I_ee_b1), Pf(I_ee_a), p_ee2, Pf(I_ee_b2), ebuf, 41);
  mlp2_plain_kernel<<<BB / 16, 128, 0, stream>>>(
      in_glob, p_eg1, Pf(I_eg_b1), Pf(I_eg_a), p_eg2, Pf(I_eg_b2), ubuf, 2);

  const int* erow = ei;
  const int* ecol = ei + NED;
  for (int l = 0; l < 3; ++l) {
    const int PB = I_proc0 + 14 * l;
    kzero<<<(NND * HH) / 256, 256, 0, stream>>>(agg, (size_t)NND * HH);
    fused_gnn_layer_kernel<<<NED / 16, 128, 0, stream>>>(
        xbuf, nullptr, erow, ecol, p_ew1[l], Pf(PB + 1), Pf(PB + 4),
        Pf(PB + 3), Pf(PB + 0), p_ew2[l], Pf(PB + 2), ebuf, agg, 3 * HH);
    fused_gnn_layer_kernel<<<NND / 16, 128, 0, stream>>>(
        nullptr, agg, nullptr, nullptr, p_nw1[l], Pf(PB + 8), Pf(PB + 11),
        Pf(PB + 10), Pf(PB + 7), p_nw2[l], Pf(PB + 9), xbuf, nullptr, 2 * HH);
  }

  seed_q_kernel<<<(NEQ * BB * HH) / 256, 256, 0, stream>>>(Pf(I_embed), qbuf);
  seed_kv_kernel<<<(NPGD * BB * HH) / 256, 256, 0, stream>>>(xbuf, kv);

  for (int t = 0; t < 2; ++t) {
    const int TB = I_tr0 + 16 * t;
    ln_kernel<<<MQ, 128, 0, stream>>>(qbuf, Pf(TB + 9), Pf(TB + 8), qn);
    ln_kernel<<<MK, 128, 0, stream>>>(kv, Pf(TB + 9), Pf(TB + 8), kn);
    wmma_gemm_kernel<false><<<dim3(MQ / 16, 1), 128, 0, stream>>>(
        qn, nullptr, 0, p_wq[t], Pf(TB + 2), qh, MQ, HH, HH, scaling, 0, 0);
    wmma_gemm_kernel<false><<<dim3(MK / 16, 1), 128, 0, stream>>>(
        kn, nullptr, 0, p_wk[t], Pf(TB + 0), kh, MK, HH, HH, 1.f, 0, 0);
    wmma_gemm_kernel<false><<<dim3(MK / 16, 1), 128, 0, stream>>>(
        kn, nullptr, 0, p_wv[t], Pf(TB + 3), vh, MK, HH, HH, 1.f, 0, 0);
    attn_score_kernel<<<dim3(13, 10, BB), 128, 0, stream>>>(qh, kh, logits);
    softmax_kernel<<<BB * NEQ, 128, 0, stream>>>(logits);
    attn_apply_kernel<<<dim3(13, 2, BB), 128, 0, stream>>>(logits, vh, aout);
    wmma_gemm_kernel<false><<<dim3(MQ / 16, 1), 128, 0, stream>>>(
        aout, nullptr, 0, p_wo[t], Pf(TB + 1), qbuf, MQ, HH, HH, 1.f, 0, 1);
    ln_kernel<<<MQ, 128, 0, stream>>>(qbuf, Pf(TB + 11), Pf(TB + 10), tmp);
    wmma_gemm_kernel<false><<<dim3(MQ / 16, 4), 128, 0, stream>>>(
        tmp, nullptr, 0, p_f1[t], Pf(TB + 6), ffh, MQ, HH, FFND, 1.f, 1, 0);
    wmma_gemm_kernel<false><<<dim3(MQ / 16, 1), 128, 0, stream>>>(
        ffh, nullptr, 0, p_f2[t], Pf(TB + 7), qbuf, MQ, FFND, HH, 1.f, 0, 1);
  }
  ln_kernel<<<MQ, 128, 0, stream>>>(qbuf, Pf(I_tr_lng), Pf(I_tr_lnb), enf);

  // decoder
  gsum_kernel<<<BB, 128, 0, stream>>>(xbuf, gsb);
  wmma_gemm_kernel<true><<<dim3(BB / 16, 1), 128, 0, stream>>>(
      ubuf, gsb, HH, p_dec, Pf(I_dec_b), grb, BB, 2 * HH, HH, 1.f, 0, 0);
  di_kernel<<<(NEQ * BB * HH) / 256, 256, 0, stream>>>(enf, grb, Pf(I_alpha),
                                                       dib);
  wmma_gemm_kernel<false><<<dim3(MQ / 16, 1), 128, 0, stream>>>(
      dib, nullptr, 0, p_ow1, Pf(I_ob1), hb, MQ, HH, HH, 1.f, 0, 0);
  ln_kernel<<<MQ, 128, 0, stream>>>(hb, Pf(I_olg), Pf(I_olb), tmp);
  prelu_kernel<<<(NEQ * BB * HH) / 256, 256, 0, stream>>>(
      tmp, Pf(I_oa), (size_t)NEQ * BB * HH);
  dos_kernel<<<MQ, 128, 0, stream>>>(tmp, Pf(I_ow2), Pf(I_ob2), out);
  (void)hipMemcpyAsync(out + BB * NEQ, xbuf, sizeof(float) * (size_t)NND * HH,
                       hipMemcpyDeviceToDevice, stream);
  spark_kernel<<<200 * BB, 128, 0, stream>>>(
      enf, Pf(I_sw), Pf(I_sb), out + BB * NEQ + (size_t)NND * HH);
}
